// GCN_63660005261514
// MI455X (gfx1250) — compile-verified, matrix operand order checked by reading, not run on hardware
//
#include <hip/hip_runtime.h>
#include <math.h>

// ---------------------------------------------------------------------------
// CDNA5 (gfx1250) GNN allconv pipeline: bf16 WMMA edge/node MLPs + atomic
// segment-max / segment-sum scatter. wave32.
// Edge kernel: one 32-edge tile per wave (2 M-tiles share B fragments),
// fully unrolled compile-time K loops, divergence-free gathered A fragments.
// ---------------------------------------------------------------------------

typedef __attribute__((ext_vector_type(16))) __bf16 v16bf;
typedef __attribute__((ext_vector_type(8)))  float  v8f;

#define WPB 4            // waves per block
#define NTHREADS 128

static __device__ __forceinline__ v16bf ldfrag(const __bf16* p) {
  return *(const v16bf*)p;
}

static __device__ __forceinline__ v8f wmma_bf(v16bf A, v16bf B, v8f C) {
  // D = A(16x32) * B(32x16) + C, f32 accumulate
  return __builtin_amdgcn_wmma_f32_16x16x32_bf16(false, A, false, B,
                                                 (short)0, C, false, false);
}

// order-preserving float -> uint key (for atomic segment-max)
static __device__ __forceinline__ unsigned fkey(float f) {
  unsigned u = __float_as_uint(f);
  return (u & 0x80000000u) ? ~u : (u | 0x80000000u);
}
static __device__ __forceinline__ float fkey_dec(unsigned k) {
  unsigned u = (k & 0x80000000u) ? (k ^ 0x80000000u) : ~k;
  return __uint_as_float(u);
}

// ---------------------------------------------------------------------------
// Edge ("msg") MLP kernel, M = 32 edges per wave.
//   u = [X[src], X[dst], ef(16) | 0pad(16)]   (K1 = 2*DIN + 32)
//   h = relu(u W1+b1); relu(h W2+b2); relu(h W3+b3); y = h W4+b4 (25 cols)
//   k = sigmoid(y0); f1 = y[1..12]*k -> seg-max(dst); f2 = y[13..24]*k -> seg-sum
// ---------------------------------------------------------------------------
struct MsgArgs {
  const __bf16* X;      // [N][DIN] bf16
  const __bf16* ef;     // [E][32]  bf16 (cols 16..31 zero)
  const int* src;
  const int* dst;
  const __bf16* W1t; const __bf16* W2t; const __bf16* W3t; const __bf16* W4t;
  const float* b1; const float* b2; const float* b3; const float* b4;
  unsigned* nf1key;
  float* nf2;
  int E;
};

// C-layout (lane=col) -> bf16 A-layout bounce tile (32x64) with bias+ReLU
static __device__ __forceinline__ void store_relu32(__bf16* act, v8f acc[2][4],
                                                    const float* bias, int lane) {
  const int col0 = lane & 15;
  const int mb = (lane >> 4) << 3;
#pragma unroll
  for (int h = 0; h < 2; ++h)
#pragma unroll
    for (int n = 0; n < 4; ++n) {
      const int col = n * 16 + col0;
      const float b = bias[col];
#pragma unroll
      for (int r = 0; r < 8; ++r) {
        float v = acc[h][n][r] + b;
        act[(h * 16 + mb + r) * 64 + col] = (__bf16)(v > 0.f ? v : 0.f);
      }
    }
}

// 32x64 act (LDS) x 64x64 Wt (global, transposed [col][k]) -> 32x64 acc
static __device__ __forceinline__ void gemm_act64_m32(const __bf16* act,
                                                      const __bf16* Wt,
                                                      v8f acc[2][4], int lane) {
  const int row = lane & 15;
  const int kh = (lane >> 4) << 4;
#pragma unroll
  for (int h = 0; h < 2; ++h)
#pragma unroll
    for (int n = 0; n < 4; ++n)
#pragma unroll
      for (int i = 0; i < 8; ++i) acc[h][n][i] = 0.f;
#pragma unroll
  for (int ks = 0; ks < 2; ++ks) {
    v16bf af0 = ldfrag(act + (row)      * 64 + (ks << 5) + kh);
    v16bf af1 = ldfrag(act + (16 + row) * 64 + (ks << 5) + kh);
#pragma unroll
    for (int n = 0; n < 4; ++n) {
      v16bf bf = ldfrag(Wt + (size_t)(n * 16 + row) * 64 + (ks << 5) + kh);
      acc[0][n] = wmma_bf(af0, bf, acc[0][n]);
      acc[1][n] = wmma_bf(af1, bf, acc[1][n]);
    }
  }
}

template <int DIN>
__global__ __launch_bounds__(NTHREADS)
void msg_mlp_kernel(MsgArgs a) {
  constexpr int K1 = 2 * DIN + 32;
  constexpr int NS = DIN / 32;       // chunks per src/dst section
  __shared__ __attribute__((aligned(32))) __bf16 actS[WPB][32 * 64];
  __shared__ __attribute__((aligned(32))) float  outS[WPB][32 * 32];
  const int lane = threadIdx.x & 31;
  const int wave = threadIdx.x >> 5;
  const int tile = blockIdx.x * WPB + wave;
  const int nTiles = (a.E + 31) >> 5;
  if (tile >= nTiles) return;  // uniform per wave

  const int row = lane & 15;
  const int kh  = (lane >> 4) << 4;  // 0 or 16
  const int e0  = tile << 5;
  int e0r = e0 + row;       if (e0r > a.E - 1) e0r = a.E - 1;
  int e1r = e0 + 16 + row;  if (e1r > a.E - 1) e1r = a.E - 1;
  const int se0 = a.src[e0r], de0 = a.dst[e0r];
  const int se1 = a.src[e1r], de1 = a.dst[e1r];
  __bf16* act = actS[wave];

  v8f acc[2][4];
#pragma unroll
  for (int h = 0; h < 2; ++h)
#pragma unroll
    for (int n = 0; n < 4; ++n)
#pragma unroll
      for (int i = 0; i < 8; ++i) acc[h][n][i] = 0.f;

  // ---- GEMM1: three divergence-free sections (src | dst | ef+pad) ----
#pragma unroll
  for (int ks = 0; ks < NS; ++ks) {
    v16bf af0 = ldfrag(a.X + (size_t)se0 * DIN + (ks << 5) + kh);
    v16bf af1 = ldfrag(a.X + (size_t)se1 * DIN + (ks << 5) + kh);
#pragma unroll
    for (int n = 0; n < 4; ++n) {
      v16bf bf = ldfrag(a.W1t + (size_t)(n * 16 + row) * K1 + (ks << 5) + kh);
      acc[0][n] = wmma_bf(af0, bf, acc[0][n]);
      acc[1][n] = wmma_bf(af1, bf, acc[1][n]);
    }
  }
#pragma unroll
  for (int ks = 0; ks < NS; ++ks) {
    v16bf af0 = ldfrag(a.X + (size_t)de0 * DIN + (ks << 5) + kh);
    v16bf af1 = ldfrag(a.X + (size_t)de1 * DIN + (ks << 5) + kh);
#pragma unroll
    for (int n = 0; n < 4; ++n) {
      v16bf bf = ldfrag(a.W1t + (size_t)(n * 16 + row) * K1 + ((NS + ks) << 5) + kh);
      acc[0][n] = wmma_bf(af0, bf, acc[0][n]);
      acc[1][n] = wmma_bf(af1, bf, acc[1][n]);
    }
  }
  {
    v16bf af0 = ldfrag(a.ef + ((size_t)e0r << 5) + kh);
    v16bf af1 = ldfrag(a.ef + ((size_t)e1r << 5) + kh);
#pragma unroll
    for (int n = 0; n < 4; ++n) {
      v16bf bf = ldfrag(a.W1t + (size_t)(n * 16 + row) * K1 + ((2 * NS) << 5) + kh);
      acc[0][n] = wmma_bf(af0, bf, acc[0][n]);
      acc[1][n] = wmma_bf(af1, bf, acc[1][n]);
    }
  }
  store_relu32(act, acc, a.b1, lane);
  gemm_act64_m32(act, a.W2t, acc, lane);
  store_relu32(act, acc, a.b2, lane);
  gemm_act64_m32(act, a.W3t, acc, lane);
  store_relu32(act, acc, a.b3, lane);

  // ---- final msg layer: 64 -> 25 (N padded to 32, Wt zero-padded) ----
  v8f o[2][2];
#pragma unroll
  for (int h = 0; h < 2; ++h)
#pragma unroll
    for (int n = 0; n < 2; ++n)
#pragma unroll
      for (int i = 0; i < 8; ++i) o[h][n][i] = 0.f;
#pragma unroll
  for (int ks = 0; ks < 2; ++ks) {
    v16bf af0 = ldfrag(act + (row)      * 64 + (ks << 5) + kh);
    v16bf af1 = ldfrag(act + (16 + row) * 64 + (ks << 5) + kh);
#pragma unroll
    for (int n = 0; n < 2; ++n) {
      v16bf bf = ldfrag(a.W4t + (size_t)(n * 16 + row) * 64 + (ks << 5) + kh);
      o[0][n] = wmma_bf(af0, bf, o[0][n]);
      o[1][n] = wmma_bf(af1, bf, o[1][n]);
    }
  }
  float* ow = outS[wave];
  {
    const int col0 = lane & 15;
    const int mb = (lane >> 4) << 3;
#pragma unroll
    for (int h = 0; h < 2; ++h)
#pragma unroll
      for (int n = 0; n < 2; ++n) {
        const int col = n * 16 + col0;
        const float b = (col < 25) ? a.b4[col] : 0.f;
#pragma unroll
        for (int r = 0; r < 8; ++r)
          ow[(h * 16 + mb + r) * 32 + col] = o[h][n][r] + b;
      }
  }

  // ---- gated scatter: 32 edges x 24 features over 32 lanes ----
  for (int idx = lane; idx < 32 * 24; idx += 32) {
    const int er = idx / 24;
    const int f  = idx - er * 24;
    const int ee = e0 + er;
    if (ee >= a.E) continue;
    const float kk = 1.f / (1.f + __expf(-ow[er * 32]));
    const float v  = ow[er * 32 + 1 + f] * kk;
    const int d = a.dst[ee];
    if (f < 12) atomicMax(&a.nf1key[(size_t)d * 12 + f], fkey(v));
    else        atomicAdd(&a.nf2[(size_t)d * 12 + (f - 12)], v);
  }
}

// ---------------------------------------------------------------------------
// Node ("red") MLP kernel over a prebuilt padded bf16 input [N][K1]
// ---------------------------------------------------------------------------
struct RedArgs {
  const __bf16* Xr;
  const __bf16* W1t; const __bf16* W2t; const __bf16* W3t; const __bf16* W4t;
  const float* b1; const float* b2; const float* b3; const float* b4;
  float* out;
  const float* resid;   // nullptr or [N][dout]
  int N; int dout;
};

static __device__ __forceinline__ void store_relu16(__bf16* act, v8f acc[4],
                                                    const float* bias, int lane) {
  const int col0 = lane & 15;
  const int mb = (lane >> 4) << 3;
#pragma unroll
  for (int n = 0; n < 4; ++n) {
    const int col = n * 16 + col0;
    const float b = bias[col];
#pragma unroll
    for (int r = 0; r < 8; ++r) {
      float v = acc[n][r] + b;
      act[(mb + r) * 64 + col] = (__bf16)(v > 0.f ? v : 0.f);
    }
  }
}

static __device__ __forceinline__ void gemm_act64_m16(const __bf16* act,
                                                      const __bf16* Wt,
                                                      v8f acc[4], int lane) {
  const int row = lane & 15;
  const int kh = (lane >> 4) << 4;
#pragma unroll
  for (int n = 0; n < 4; ++n)
#pragma unroll
    for (int i = 0; i < 8; ++i) acc[n][i] = 0.f;
#pragma unroll
  for (int ks = 0; ks < 2; ++ks) {
    v16bf af = ldfrag(act + row * 64 + (ks << 5) + kh);
#pragma unroll
    for (int n = 0; n < 4; ++n) {
      v16bf bf = ldfrag(Wt + (size_t)(n * 16 + row) * 64 + (ks << 5) + kh);
      acc[n] = wmma_bf(af, bf, acc[n]);
    }
  }
}

template <int K1>
__global__ __launch_bounds__(NTHREADS)
void red_mlp_kernel(RedArgs a) {
  constexpr int NK = K1 / 32;
  __shared__ __attribute__((aligned(32))) __bf16 actS[WPB][16 * 64];
  const int lane = threadIdx.x & 31;
  const int wave = threadIdx.x >> 5;
  const int tile = blockIdx.x * WPB + wave;
  const int nTiles = (a.N + 15) >> 4;
  if (tile >= nTiles) return;

  const int row = lane & 15;
  const int kh = (lane >> 4) << 4;
  const int n0 = tile << 4;
  int node = n0 + row;
  if (node > a.N - 1) node = a.N - 1;
  __bf16* act = actS[wave];

  v8f acc[4];
#pragma unroll
  for (int n = 0; n < 4; ++n)
#pragma unroll
    for (int i = 0; i < 8; ++i) acc[n][i] = 0.f;

#pragma unroll
  for (int ks = 0; ks < NK; ++ks) {
    v16bf af = ldfrag(a.Xr + (size_t)node * K1 + (ks << 5) + kh);
#pragma unroll
    for (int n = 0; n < 4; ++n) {
      v16bf bf = ldfrag(a.W1t + (size_t)(n * 16 + row) * K1 + (ks << 5) + kh);
      acc[n] = wmma_bf(af, bf, acc[n]);
    }
  }
  store_relu16(act, acc, a.b1, lane);
  gemm_act64_m16(act, a.W2t, acc, lane);
  store_relu16(act, acc, a.b2, lane);
  gemm_act64_m16(act, a.W3t, acc, lane);
  store_relu16(act, acc, a.b3, lane);
  gemm_act64_m16(act, a.W4t, acc, lane);   // W4t zero-padded to [64][64]

  const int col0 = lane & 15;
  const int mb = (lane >> 4) << 3;
#pragma unroll
  for (int n = 0; n < 4; ++n) {
    const int col = n * 16 + col0;
    if (col >= a.dout) continue;
    const float b = a.b4[col];
#pragma unroll
    for (int r = 0; r < 8; ++r) {
      const int nd = n0 + mb + r;
      if (nd >= a.N) continue;
      float v = acc[n][r] + b;
      if (a.resid) v += a.resid[(size_t)nd * a.dout + col];
      a.out[(size_t)nd * a.dout + col] = v;
    }
  }
}

// ---------------------------------------------------------------------------
// Prep / utility kernels
// ---------------------------------------------------------------------------
__global__ void k_f32_to_bf16(const float* x, __bf16* y, int n) {
  int i = blockIdx.x * blockDim.x + threadIdx.x;
  if (i < n) y[i] = (__bf16)x[i];
}

// ef [E][16] f32 -> efb [E][32] bf16 (cols 16..31 = 0)
__global__ void k_build_ef(const float* ef, __bf16* efb, int E) {
  int i = blockIdx.x * blockDim.x + threadIdx.x;
  if (i >= E * 32) return;
  int e = i >> 5;
  int c = i & 31;
  efb[i] = (c < 16) ? (__bf16)ef[(size_t)e * 16 + c] : (__bf16)0.0f;
}

// W [K][Nout] f32 -> Wt [Npad][Kpad] bf16 transposed, zero-padded
__global__ void k_convert_wt(const float* W, __bf16* Wt, int K, int Nout,
                             int Kpad, int Npad) {
  int i = blockIdx.x * blockDim.x + threadIdx.x;
  if (i >= Kpad * Npad) return;
  int n = i / Kpad;
  int k = i - n * Kpad;
  float v = (k < K && n < Nout) ? W[(size_t)k * Nout + n] : 0.f;
  Wt[i] = (__bf16)v;
}

__global__ void k_init_seg(unsigned* key, float* sum, int n) {
  int i = blockIdx.x * blockDim.x + threadIdx.x;
  if (i < n) { key[i] = 0x007FFFFFu; /* fkey(-inf) */ sum[i] = 0.f; }
}

// Build padded red input [N][din+32] = [X | nf1(max,-inf->0) | nf2(sum) | 0]
__global__ void k_build_xr(const __bf16* X, const unsigned* key, const float* sum,
                           __bf16* Xr, int N, int din) {
  const int Kr = din + 32;
  int i = blockIdx.x * blockDim.x + threadIdx.x;
  if (i >= N * Kr) return;
  int node = i / Kr;
  int c = i - node * Kr;
  if (c < din) { Xr[i] = X[(size_t)node * din + c]; return; }
  float v;
  if (c < din + 12) {
    float f = fkey_dec(key[(size_t)node * 12 + (c - din)]);
    v = (f >= -3.0e38f) ? f : 0.f;   // -inf (isolated node) -> 0
  } else if (c < din + 24) {
    v = sum[(size_t)node * 12 + (c - din - 12)];
  } else {
    v = 0.f;
  }
  Xr[i] = (__bf16)v;
}

// layer-1 input: [N][128] = [bf16(x1) | nf_bf16]
__global__ void k_build_xc1(const float* x1, const __bf16* nfb, __bf16* Xc, int N) {
  int i = blockIdx.x * blockDim.x + threadIdx.x;
  if (i >= N * 128) return;
  int node = i >> 7;
  int c = i & 127;
  Xc[i] = (c < 64) ? (__bf16)x1[(size_t)node * 64 + c]
                   : nfb[(size_t)node * 64 + (c - 64)];
}

// ---------------------------------------------------------------------------
// Host
// ---------------------------------------------------------------------------
extern "C" void kernel_launch(void* const* d_in, const int* in_sizes, int n_in,
                              void* d_out, int out_size, void* d_ws, size_t ws_size,
                              hipStream_t stream) {
  const int N = in_sizes[0] / 64;
  const int E = in_sizes[1] / 16;
  const float* nf = (const float*)d_in[0];
  const float* ef = (const float*)d_in[1];
  // params: 3 layers x {msg: (W,b)x4, red: (W,b)x4} in dict/insertion order
  const float* P[48];
  for (int i = 0; i < 48; ++i) P[i] = (const float*)d_in[2 + i];
  const int* src = (const int*)d_in[n_in - 2];
  const int* dst = (const int*)d_in[n_in - 1];

  char* ws = (char*)d_ws;
  size_t off = 0;
  auto alloc = [&](size_t bytes) -> void* {
    void* p = ws + off;
    off = (off + bytes + 255) & ~(size_t)255;
    return p;
  };

  __bf16* X0  = (__bf16*)alloc((size_t)N * 64 * 2);
  __bf16* Xc1 = (__bf16*)alloc((size_t)N * 128 * 2);
  __bf16* X2  = (__bf16*)alloc((size_t)N * 64 * 2);
  __bf16* efb = (__bf16*)alloc((size_t)E * 32 * 2);
  unsigned* nf1key = (unsigned*)alloc((size_t)N * 12 * 4);
  float* nf2 = (float*)alloc((size_t)N * 12 * 4);
  __bf16* Xr = (__bf16*)alloc((size_t)N * 160 * 2);  // max Kr
  float* x1 = (float*)alloc((size_t)N * 64 * 4);
  float* x2 = (float*)alloc((size_t)N * 64 * 4);

  const int dinL[3]  = {64, 128, 64};
  const int doutL[3] = {64, 64, 8};
  __bf16 *mW1[3], *mW2[3], *mW3[3], *mW4[3];
  __bf16 *rW1[3], *rW2[3], *rW3[3], *rW4[3];
  for (int l = 0; l < 3; ++l) {
    const int K1m = 2 * dinL[l] + 32;
    const int K1r = dinL[l] + 32;
    mW1[l] = (__bf16*)alloc((size_t)64 * K1m * 2);
    mW2[l] = (__bf16*)alloc((size_t)64 * 64 * 2);
    mW3[l] = (__bf16*)alloc((size_t)64 * 64 * 2);
    mW4[l] = (__bf16*)alloc((size_t)32 * 64 * 2);
    rW1[l] = (__bf16*)alloc((size_t)64 * K1r * 2);
    rW2[l] = (__bf16*)alloc((size_t)64 * 64 * 2);
    rW3[l] = (__bf16*)alloc((size_t)64 * 64 * 2);
    rW4[l] = (__bf16*)alloc((size_t)64 * 64 * 2);
  }

  auto grid1 = [](long long n) { return (unsigned)((n + 255) / 256); };

  k_f32_to_bf16<<<grid1((long long)N * 64), 256, 0, stream>>>(nf, X0, N * 64);
  k_build_ef<<<grid1((long long)E * 32), 256, 0, stream>>>(ef, efb, E);

  auto cvt = [&](const float* W, __bf16* Wt, int K, int Nout, int Kpad, int Npad) {
    k_convert_wt<<<grid1((long long)Kpad * Npad), 256, 0, stream>>>(W, Wt, K, Nout,
                                                                    Kpad, Npad);
  };
  for (int l = 0; l < 3; ++l) {
    const int base = l * 16;
    const int K1m = 2 * dinL[l] + 32;
    const int K1r = dinL[l] + 32;
    cvt(P[base + 0],  mW1[l], 2 * dinL[l] + 16, 64, K1m, 64);
    cvt(P[base + 2],  mW2[l], 64, 64, 64, 64);
    cvt(P[base + 4],  mW3[l], 64, 64, 64, 64);
    cvt(P[base + 6],  mW4[l], 64, 25, 64, 32);
    cvt(P[base + 8],  rW1[l], dinL[l] + 24, 64, K1r, 64);
    cvt(P[base + 10], rW2[l], 64, 64, 64, 64);
    cvt(P[base + 12], rW3[l], 64, 64, 64, 64);
    cvt(P[base + 14], rW4[l], 64, doutL[l], 64, 64);
  }

  auto runLayer = [&](int l, const __bf16* Xin, float* xout, const float* resid) {
    const int base = l * 16;
    const int din = dinL[l], dout = doutL[l];
    const int K1r = din + 32;
    k_init_seg<<<grid1((long long)N * 12), 256, 0, stream>>>(nf1key, nf2, N * 12);

    MsgArgs ma;
    ma.X = Xin; ma.ef = efb; ma.src = src; ma.dst = dst;
    ma.W1t = mW1[l]; ma.W2t = mW2[l]; ma.W3t = mW3[l]; ma.W4t = mW4[l];
    ma.b1 = P[base + 1]; ma.b2 = P[base + 3]; ma.b3 = P[base + 5]; ma.b4 = P[base + 7];
    ma.nf1key = nf1key; ma.nf2 = nf2;
    ma.E = E;
    const int eTiles = (E + 31) / 32;
    const int eBlocks = (eTiles + WPB - 1) / WPB;
    if (din == 64)
      msg_mlp_kernel<64><<<eBlocks, NTHREADS, 0, stream>>>(ma);
    else
      msg_mlp_kernel<128><<<eBlocks, NTHREADS, 0, stream>>>(ma);

    k_build_xr<<<grid1((long long)N * K1r), 256, 0, stream>>>(Xin, nf1key, nf2, Xr,
                                                              N, din);

    RedArgs ra;
    ra.Xr = Xr;
    ra.W1t = rW1[l]; ra.W2t = rW2[l]; ra.W3t = rW3[l]; ra.W4t = rW4[l];
    ra.b1 = P[base + 9]; ra.b2 = P[base + 11]; ra.b3 = P[base + 13]; ra.b4 = P[base + 15];
    ra.out = xout; ra.resid = resid;
    ra.N = N; ra.dout = dout;
    const int nTiles = (N + 15) / 16;
    const int nBlocks = (nTiles + WPB - 1) / WPB;
    if (K1r == 96)
      red_mlp_kernel<96><<<nBlocks, NTHREADS, 0, stream>>>(ra);
    else
      red_mlp_kernel<160><<<nBlocks, NTHREADS, 0, stream>>>(ra);
  };

  runLayer(0, X0, x1, nullptr);
  k_build_xc1<<<grid1((long long)N * 128), 256, 0, stream>>>(x1, X0, Xc1, N);
  runLayer(1, Xc1, x2, x1);
  k_f32_to_bf16<<<grid1((long long)N * 64), 256, 0, stream>>>(x2, X2, N * 64);
  runLayer(2, X2, (float*)d_out, nullptr);

  (void)out_size; (void)ws_size;
}
